// DCNV3_YOLO_25074019074533
// MI455X (gfx1250) — compile-verified, matrix operand order checked by reading, not run on hardware
//
#include <hip/hip_runtime.h>

// ---------------- problem constants ----------------
#define NB   4
#define C    128
#define H    96
#define W    96
#define HW   (H * W)            // 9216
#define NPIX (NB * HW)          // 36864
#define G    4
#define GC   32
#define P    9
#define EPSF 1e-5f

typedef __attribute__((ext_vector_type(16))) __bf16 v16bf;
typedef __attribute__((ext_vector_type(8)))  __bf16 v8bf;
typedef __attribute__((ext_vector_type(8)))  float  v8f;
typedef __attribute__((ext_vector_type(4)))  unsigned int v4u;
typedef __attribute__((ext_vector_type(8)))  int v8i;
typedef __attribute__((ext_vector_type(4)))  int v4i;

// ---------------------------------------------------------------------
// TDM helper: 1-D DMA of `ndw` dwords from global -> LDS (D# per ISA ch.8)
// group0: count=1, lds_addr, 57-bit global addr, type=2
// group1: data_size=4B, tensor_dim0=tile_dim0=ndw, dims1/2 trivial
// toolchain: clang-23 6-arg builtin (g0, g1, g2, g3, g4, cpol)
// ---------------------------------------------------------------------
__device__ __forceinline__ void tdm_load_1d(unsigned int lds_addr,
                                            const void* gptr,
                                            unsigned int ndw) {
  unsigned long long ga = (unsigned long long)(uintptr_t)gptr;
  v4u g0 = { 1u,                                   // count=1, user mode
             lds_addr,                             // bits 63:32  lds_addr
             (unsigned int)(ga & 0xFFFFFFFFu),     // bits 95:64  gaddr lo
             (unsigned int)((ga >> 32) & 0x01FFFFFFu) | (2u << 30) }; // type=2
  v8i g1 = { (int)(2u << 16),                      // data_size = 2 (4 bytes)
             (int)((ndw & 0xFFFFu) << 16),         // tensor_dim0 lo16 @ bits 63:48
             (int)((ndw >> 16) | (1u << 16)),      // tensor_dim0 hi16 | tensor_dim1=1
             (int)((ndw & 0xFFFFu) << 16),         // tile_dim0 @ bits 127:112
             1,                                    // tile_dim1 = 1
             (int)ndw,                             // tensor_dim0_stride lo32
             0, 0 };
  v4i z4 = { 0, 0, 0, 0 };
  v8i z8 = { 0, 0, 0, 0, 0, 0, 0, 0 };
  __builtin_amdgcn_tensor_load_to_lds(g0, g1, z4, z4, z8, 0);
}

// fragment K index helper (16-bit A/B layout, wave32)
__device__ __forceinline__ int frag_k(int lane, int i) {
  const int kgrp = (lane >> 4) << 3;               // 0 or 8
  return (i < 8) ? (kgrp + i) : (16 + kgrp + (i - 8));
}

// =====================================================================
// Kernel 0: pack weights (f32 (K,Ncols) row-major) -> bf16 fragment order
// layout: pb[((tn*4 + kk)*32 + lane)*16 + i]
// supports a concatenated pair (w0 | w1) with zero padding past n0+n1
// =====================================================================
__global__ void k_pack(const float* __restrict__ w0, int n0,
                       const float* __restrict__ w1, int n1,
                       int ntiles, __bf16* __restrict__ pb) {
  const int idx = blockIdx.x * 256 + threadIdx.x;
  if (idx >= ntiles * 2048) return;
  const int i    = idx & 15;
  const int lane = (idx >> 4) & 31;
  const int kk   = (idx >> 9) & 3;
  const int tn   = idx >> 11;
  const int K    = kk * 32 + frag_k(lane, i);
  const int col  = tn * 16 + (lane & 15);
  float v = 0.f;
  if (col < n0)            v = w0[K * n0 + col];
  else if (col < n0 + n1)  v = w1[K * n1 + (col - n0)];
  pb[idx] = (__bf16)v;
}

// =====================================================================
// Kernel 1: depthwise 3x3 + bias -> LayerNorm(C) -> exact GELU -> x1 bf16
// also emits x_nhwc in bf16 (contiguous A operand for input_proj)
// =====================================================================
__global__ void k_dw_ln_gelu(const float* __restrict__ x,
                             const float* __restrict__ dw_w,
                             const float* __restrict__ dw_b,
                             const float* __restrict__ ln_g,
                             const float* __restrict__ ln_b,
                             __bf16* __restrict__ x1,
                             __bf16* __restrict__ xb) {
  const int m = blockIdx.x;            // pixel index
  const int c = threadIdx.x;           // channel 0..127
  const int n_img = m / HW;
  const int pp    = m - n_img * HW;
  const int h = pp / W;
  const int w = pp - h * W;

  const float* xc = x + (size_t)(n_img * C + c) * HW;
  float s = 0.f;
  float center = 0.f;
#pragma unroll
  for (int ky = 0; ky < 3; ++ky) {
#pragma unroll
    for (int kx = 0; kx < 3; ++kx) {
      const int yy = h + ky - 1;
      const int xx = w + kx - 1;
      const float v = (yy >= 0 && yy < H && xx >= 0 && xx < W) ? xc[yy * W + xx] : 0.f;
      if (ky == 1 && kx == 1) center = v;
      s += v * dw_w[c * 9 + ky * 3 + kx];
    }
  }
  s += dw_b[c];
  xb[(size_t)m * C + c] = (__bf16)center;        // x in NHWC bf16

  float sum = s, sumsq = s * s;
#pragma unroll
  for (int o = 16; o > 0; o >>= 1) {
    sum   += __shfl_xor(sum,   o, 32);
    sumsq += __shfl_xor(sumsq, o, 32);
  }
  __shared__ float s1[4], s2[4];
  const int warp = threadIdx.x >> 5;
  const int lane = threadIdx.x & 31;
  if (lane == 0) { s1[warp] = sum; s2[warp] = sumsq; }
  __syncthreads();
  const float tsum = s1[0] + s1[1] + s1[2] + s1[3];
  const float tsq  = s2[0] + s2[1] + s2[2] + s2[3];
  const float mu  = tsum * (1.f / 128.f);
  const float var = tsq * (1.f / 128.f) - mu * mu;

  const float xn = (s - mu) * rsqrtf(var + EPSF) * ln_g[c] + ln_b[c];
  const float gl = 0.5f * xn * (1.f + erff(xn * 0.70710678118654752f));
  x1[(size_t)m * C + c] = (__bf16)gl;
}

// ---------------------------------------------------------------------
// shared GEMM core: one block = one 16-row M stripe; wave `tn` computes
// the 16x16 tile at column tile tn. A (16x128 bf16, 4 KB) and packed W
// (NT*4 KB) are DMA'd into LDS by wave 0 via the Tensor Data Mover.
// ---------------------------------------------------------------------
template <int NT>
__device__ __forceinline__ v8f gemm_tile(const __bf16* __restrict__ aSrc,
                                         const __bf16* __restrict__ wPack,
                                         __bf16* ldsA, __bf16* ldsW) {
  const int lane = threadIdx.x & 31;
  const int warp = threadIdx.x >> 5;
  const int tn   = warp;

  if (warp == 0) {
    tdm_load_1d((unsigned int)(uintptr_t)ldsA, aSrc,  (16 * C * 2) / 4);
    tdm_load_1d((unsigned int)(uintptr_t)ldsW, wPack, (NT * 2048 * 2) / 4);
    __builtin_amdgcn_s_wait_tensorcnt(0);
  }
  __syncthreads();

  const int kgrp = (lane >> 4) << 3;
  const __bf16* arow = ldsA + (lane & 15) * C;

  v8f acc = {};
#pragma unroll
  for (int kk = 0; kk < 4; ++kk) {
    const v8bf alo = *(const v8bf*)(arow + kk * 32 + kgrp);
    const v8bf ahi = *(const v8bf*)(arow + kk * 32 + 16 + kgrp);
    const v16bf a = __builtin_shufflevector(alo, ahi, 0, 1, 2, 3, 4, 5, 6, 7,
                                            8, 9, 10, 11, 12, 13, 14, 15);
    const v16bf b = *(const v16bf*)(ldsW + ((tn * 4 + kk) * 32 + lane) * 16);
    acc = __builtin_amdgcn_wmma_f32_16x16x32_bf16(false, a, false, b,
                                                  (short)0, acc, false, false);
  }
  return acc;
}

// =====================================================================
// Kernel 2: input projection  (xb @ w_in + b_in) -> x_proj f32 NHWC
// =====================================================================
__global__ void k_input_proj(const __bf16* __restrict__ xb,
                             const __bf16* __restrict__ pw_in,
                             const float* __restrict__ b_in,
                             float* __restrict__ xproj) {
  __shared__ __bf16 ldsA[16 * C];
  __shared__ __bf16 ldsW[8 * 2048];
  const int lane = threadIdx.x & 31;
  const int tn   = threadIdx.x >> 5;
  const int tm   = blockIdx.x;

  v8f acc = gemm_tile<8>(xb + (size_t)tm * 16 * C, pw_in, ldsA, ldsW);

  const int col   = tn * 16 + (lane & 15);
  const int mBase = tm * 16 + ((lane >> 4) << 3);
#pragma unroll
  for (int v = 0; v < 8; ++v)
    xproj[(size_t)(mBase + v) * C + col] = acc[v] + b_in[col];
}

// =====================================================================
// Kernel 3: offset/mask GEMM  (x1 @ [w_off|w_mask]) ; 7 padded N tiles
// =====================================================================
__global__ void k_offmask(const __bf16* __restrict__ x1,
                          const __bf16* __restrict__ pw_om,
                          const float* __restrict__ b_off,
                          const float* __restrict__ b_mask,
                          float* __restrict__ offb,
                          float* __restrict__ mlog) {
  __shared__ __bf16 ldsA[16 * C];
  __shared__ __bf16 ldsW[7 * 2048];
  const int lane = threadIdx.x & 31;
  const int tn   = threadIdx.x >> 5;
  const int tm   = blockIdx.x;

  v8f acc = gemm_tile<7>(x1 + (size_t)tm * 16 * C, pw_om, ldsA, ldsW);

  const int col   = tn * 16 + (lane & 15);
  const int mBase = tm * 16 + ((lane >> 4) << 3);
#pragma unroll
  for (int v = 0; v < 8; ++v) {
    const int mm = mBase + v;
    if (col < 72)       offb[(size_t)mm * 72 + col]        = acc[v] + b_off[col];
    else if (col < 108) mlog[(size_t)mm * 36 + (col - 72)] = acc[v] + b_mask[col - 72];
  }
}

// =====================================================================
// Kernel 4: DCNv3 core. One wave per (pixel, group); lane == group channel.
// =====================================================================
__global__ void k_dcn(const float* __restrict__ xproj,
                      const float* __restrict__ offb,
                      const float* __restrict__ mlog,
                      __bf16* __restrict__ dout) {
  const int lane = threadIdx.x & 31;
  const int warp = threadIdx.x >> 5;
  const int wg = blockIdx.x * 8 + warp;
  const int g = wg & 3;
  const int m = wg >> 2;
  const int n_img = m / HW;
  const int pp    = m - n_img * HW;
  const int h = pp / W;
  const int w = pp - h * W;

  const float* off = offb + (size_t)m * 72 + g * 18;
  const float* ml  = mlog + (size_t)m * 36 + g * 9;

  float l[P];
  float mx = -3.0e38f;
#pragma unroll
  for (int p = 0; p < P; ++p) { l[p] = ml[p]; mx = fmaxf(mx, l[p]); }
  float ssum = 0.f;
#pragma unroll
  for (int p = 0; p < P; ++p) { l[p] = __expf(l[p] - mx); ssum += l[p]; }
  const float sinv = 1.f / ssum;

  const int cbase = g * GC + lane;
  float acc = 0.f;
#pragma unroll
  for (int p = 0; p < P; ++p) {
    const float gx = (float)(p / 3) - 1.f;       // torch meshgrid(ij)
    const float gy = (float)(p % 3) - 1.f;
    const float ix = (float)w + 1.f + gx + off[p * 2 + 0];
    const float iy = (float)h + 1.f + gy + off[p * 2 + 1];
    const float x0f = floorf(ix), y0f = floorf(iy);
    const float wx1 = ix - x0f,  wy1 = iy - y0f;
    const float wx0 = 1.f - wx1, wy0 = 1.f - wy1;
    const int x0 = (int)x0f, y0 = (int)y0f;

    float v00 = 0.f, v01 = 0.f, v10 = 0.f, v11 = 0.f;
#define TAP(yi, xi, dst)                                                        \
    if ((yi) >= 1 && (yi) < (H + 1) && (xi) >= 1 && (xi) < (W + 1)) {           \
      dst = xproj[((size_t)(n_img * HW + ((yi) - 1) * W + ((xi) - 1))) * C + cbase]; \
    }
    TAP(y0,     x0,     v00)
    TAP(y0,     x0 + 1, v01)
    TAP(y0 + 1, x0,     v10)
    TAP(y0 + 1, x0 + 1, v11)
#undef TAP
    const float bil = v00 * wy0 * wx0 + v01 * wy0 * wx1 +
                      v10 * wy1 * wx0 + v11 * wy1 * wx1;
    acc += l[p] * sinv * bil;
  }
  dout[(size_t)m * C + cbase] = (__bf16)acc;
}

// =====================================================================
// Kernel 5: output projection + BatchNorm(eval) + SiLU -> NCHW f32
// =====================================================================
__global__ void k_out_proj(const __bf16* __restrict__ dcnout,
                           const __bf16* __restrict__ pw_out,
                           const float* __restrict__ b_out,
                           const float* __restrict__ bn_g,
                           const float* __restrict__ bn_b,
                           const float* __restrict__ bn_mean,
                           const float* __restrict__ bn_var,
                           float* __restrict__ out) {
  __shared__ __bf16 ldsA[16 * C];
  __shared__ __bf16 ldsW[8 * 2048];
  const int lane = threadIdx.x & 31;
  const int tn   = threadIdx.x >> 5;
  const int tm   = blockIdx.x;

  v8f acc = gemm_tile<8>(dcnout + (size_t)tm * 16 * C, pw_out, ldsA, ldsW);

  const int col = tn * 16 + (lane & 15);
  const float scale = rsqrtf(bn_var[col] + EPSF) * bn_g[col];
  const float mean  = bn_mean[col];
  const float beta  = bn_b[col];
  const float bias  = b_out[col];
  const int mBase = tm * 16 + ((lane >> 4) << 3);
#pragma unroll
  for (int v = 0; v < 8; ++v) {
    const int mm = mBase + v;
    const int ni = mm / HW;
    const int pi = mm - ni * HW;
    float y = acc[v] + bias;
    y = (y - mean) * scale + beta;
    out[((size_t)ni * C + col) * HW + pi] = y / (1.f + __expf(-y));
  }
}

// =====================================================================
// host-side launcher
// =====================================================================
extern "C" void kernel_launch(void* const* d_in, const int* in_sizes, int n_in,
                              void* d_out, int out_size, void* d_ws, size_t ws_size,
                              hipStream_t stream) {
  const float* x       = (const float*)d_in[0];
  const float* dw_w    = (const float*)d_in[1];
  const float* dw_b    = (const float*)d_in[2];
  const float* ln_g    = (const float*)d_in[3];
  const float* ln_b    = (const float*)d_in[4];
  const float* w_off   = (const float*)d_in[5];
  const float* b_off   = (const float*)d_in[6];
  const float* w_mask  = (const float*)d_in[7];
  const float* b_mask  = (const float*)d_in[8];
  const float* w_in    = (const float*)d_in[9];
  const float* b_in    = (const float*)d_in[10];
  const float* w_out   = (const float*)d_in[11];
  const float* b_out   = (const float*)d_in[12];
  const float* bn_g    = (const float*)d_in[13];
  const float* bn_b    = (const float*)d_in[14];
  const float* bn_mean = (const float*)d_in[15];
  const float* bn_var  = (const float*)d_in[16];
  float* out = (float*)d_out;

  // workspace layout
  char* ws = (char*)d_ws;
  size_t o = 0;
  __bf16* x1     = (__bf16*)(ws + o); o += (size_t)NPIX * C * 2;   // 9.4 MB
  __bf16* xb     = (__bf16*)(ws + o); o += (size_t)NPIX * C * 2;   // 9.4 MB
  float*  xproj  = (float*) (ws + o); o += (size_t)NPIX * C * 4;   // 18.9 MB
  float*  offb   = (float*) (ws + o); o += (size_t)NPIX * 72 * 4;  // 10.6 MB
  float*  mlog   = (float*) (ws + o); o += (size_t)NPIX * 36 * 4;  // 5.3 MB
  __bf16* dcnout = (__bf16*)(ws + o); o += (size_t)NPIX * C * 2;   // 9.4 MB
  __bf16* pw_in  = (__bf16*)(ws + o); o += 8 * 2048 * 2;           // 32 KB
  __bf16* pw_out = (__bf16*)(ws + o); o += 8 * 2048 * 2;           // 32 KB
  __bf16* pw_om  = (__bf16*)(ws + o); o += 7 * 2048 * 2;           // 28 KB

  // 0) pack weights into bf16 WMMA fragment order
  k_pack<<<(8 * 2048 + 255) / 256, 256, 0, stream>>>(w_in,  C, nullptr, 0, 8, pw_in);
  k_pack<<<(8 * 2048 + 255) / 256, 256, 0, stream>>>(w_out, C, nullptr, 0, 8, pw_out);
  k_pack<<<(7 * 2048 + 255) / 256, 256, 0, stream>>>(w_off, 72, w_mask, 36, 7, pw_om);

  // 1) depthwise + LN + GELU; also x -> NHWC bf16
  k_dw_ln_gelu<<<NPIX, 128, 0, stream>>>(x, dw_w, dw_b, ln_g, ln_b, x1, xb);

  // 2) input projection: one block per 16-row stripe, 8 waves = 8 N tiles
  k_input_proj<<<NPIX / 16, 256, 0, stream>>>(xb, pw_in, b_in, xproj);

  // 3) offset/mask GEMM: 7 waves = 7 padded N tiles
  k_offmask<<<NPIX / 16, 224, 0, stream>>>(x1, pw_om, b_off, b_mask, offb, mlog);

  // 4) DCN sampling core: one wave per (pixel, group)
  k_dcn<<<NPIX * G / 8, 256, 0, stream>>>(xproj, offb, mlog, dcnout);

  // 5) output projection + BN + SiLU -> NCHW
  k_out_proj<<<NPIX / 16, 256, 0, stream>>>(dcnout, pw_out, b_out,
                                            bn_g, bn_b, bn_mean, bn_var, out);
}
